// DeformableAttention_70145405878825
// MI455X (gfx1250) — compile-verified, matrix operand order checked by reading, not run on hardware
//
#include <hip/hip_runtime.h>
#include <hip/hip_bf16.h>
#include <stdint.h>

// Problem constants (from reference)
#define E_DIM  1024
#define H_DIM  16
#define P_DIM  4
#define LQ_DIM 2048
#define LV_DIM 4096
#define BZ_DIM 4
#define HD_DIM 64

typedef __attribute__((ext_vector_type(16))) __bf16 v16bf;
typedef __attribute__((ext_vector_type(8)))  __bf16 v8bf;
typedef __attribute__((ext_vector_type(8)))  float  v8f;

// ---------------------------------------------------------------------------
// Streaming f32 -> bf16 pack: 8 elements per thread, b128-in / b128-out.
// count must be a multiple of 2048 (all our buffers are).
// ---------------------------------------------------------------------------
__global__ __launch_bounds__(256) void pack_bf16_kernel(
    const float* __restrict__ src, __bf16* __restrict__ dst)
{
    const size_t i = ((size_t)blockIdx.x * 256 + threadIdx.x) * 8;
    const float4* s = (const float4*)(src + i);
    float4 a = s[0], b = s[1];
    v8bf o;
    o[0] = (__bf16)a.x; o[1] = (__bf16)a.y; o[2] = (__bf16)a.z; o[3] = (__bf16)a.w;
    o[4] = (__bf16)b.x; o[5] = (__bf16)b.y; o[6] = (__bf16)b.z; o[7] = (__bf16)b.w;
    *(v8bf*)(dst + i) = o;
}

// Load one WMMA bf16 operand fragment for this lane from a packed bf16
// row-major [rows, K] matrix.  CDNA5 16-bit A (16x32) layout: lane holds row
// (lane&15); with half = lane>>4: elements 0..7 -> K = half*8 + 0..7,
// elements 8..15 -> K = 16 + half*8 + 0..7.  Two b128 loads, no conversion.
static __device__ __forceinline__ v16bf load_frag_bf(
    const __bf16* __restrict__ base, int K, int row, int k0, int half)
{
    const __bf16* p = base + (size_t)row * K + k0 + half * 8;
    v8bf lo = *(const v8bf*)(p);
    v8bf hi = *(const v8bf*)(p + 16);
    return __builtin_shufflevector(lo, hi, 0, 1, 2, 3, 4, 5, 6, 7,
                                   8, 9, 10, 11, 12, 13, 14, 15);
}

// ---------------------------------------------------------------------------
// Generic GEMM:  C[M,N] = A[M,K] @ B[N,K]^T + bias[N]   (A, B packed bf16)
// Block = 256 threads = 8 waves (4x2).  Wave macro-tile 32x32 (2x2 WMMA
// tiles, 4 wmma per K-step sharing 2 A-frags and 2 B-frags).
// Block tile 128(M) x 64(N).  M%128==0, N%64==0 or N==64, K%32==0.
// ---------------------------------------------------------------------------
__global__ __launch_bounds__(256) void gemm_bias_wmma(
    const __bf16* __restrict__ A, const __bf16* __restrict__ Bw,
    const float* __restrict__ bias, float* __restrict__ C,
    int M, int N, int K)
{
    const int tid  = threadIdx.x;
    const int wave = tid >> 5, lane = tid & 31;
    const int tm   = blockIdx.x * 128 + (wave & 3) * 32;
    const int tn   = blockIdx.y * 64  + (wave >> 2) * 32;
    const int half = lane >> 4, l16 = lane & 15;

    const int am0 = tm + l16, am1 = tm + 16 + l16;
    const int bn0 = tn + l16, bn1 = tn + 16 + l16;

    v8f c00 = {}, c01 = {}, c10 = {}, c11 = {};
    for (int k0 = 0; k0 < K; k0 += 32) {
        if (k0 + 32 < K) {
            __builtin_prefetch(A  + (size_t)am0 * K + k0 + 32, 0, 1);
            __builtin_prefetch(Bw + (size_t)bn0 * K + k0 + 32, 0, 1);
        }
        v16bf a0 = load_frag_bf(A,  K, am0, k0, half);
        v16bf a1 = load_frag_bf(A,  K, am1, k0, half);
        v16bf b0 = load_frag_bf(Bw, K, bn0, k0, half);
        v16bf b1 = load_frag_bf(Bw, K, bn1, k0, half);
        c00 = __builtin_amdgcn_wmma_f32_16x16x32_bf16(false, a0, false, b0, (short)0, c00, false, false);
        c01 = __builtin_amdgcn_wmma_f32_16x16x32_bf16(false, a0, false, b1, (short)0, c01, false, false);
        c10 = __builtin_amdgcn_wmma_f32_16x16x32_bf16(false, a1, false, b0, (short)0, c10, false, false);
        c11 = __builtin_amdgcn_wmma_f32_16x16x32_bf16(false, a1, false, b1, (short)0, c11, false, false);
    }

    // C/D layout: element (m = v + 8*half, n = lane%16) in acc[v]
    const int gn0 = tn + l16, gn1 = tn + 16 + l16;
    const float bv0 = bias[gn0], bv1 = bias[gn1];
#pragma unroll
    for (int v = 0; v < 8; ++v) {
        const int gm0 = tm + v + 8 * half;
        const int gm1 = gm0 + 16;
        C[(size_t)gm0 * N + gn0] = c00[v] + bv0;
        C[(size_t)gm0 * N + gn1] = c01[v] + bv1;
        C[(size_t)gm1 * N + gn0] = c10[v] + bv0;
        C[(size_t)gm1 * N + gn1] = c11[v] + bv1;
    }
}

// ---------------------------------------------------------------------------
// Value projection GEMM with fused mask + scatter into sampling layout:
//   A = value_bf [LV*BZ, E] (row r = lv*BZ + b), B = Wv_bf [E, E]
//   Vs[(b*H + h) * LV * HD + lv * HD + hd] = masked ? 0 : (A@B^T + b_value)
// ---------------------------------------------------------------------------
__global__ __launch_bounds__(256) void gemm_value_wmma(
    const __bf16* __restrict__ A, const __bf16* __restrict__ Bw,
    const float* __restrict__ bias, const unsigned char* __restrict__ mask,
    float* __restrict__ Vs)
{
    const int K = E_DIM;
    const int tid  = threadIdx.x;
    const int wave = tid >> 5, lane = tid & 31;
    const int tm   = blockIdx.x * 128 + (wave & 3) * 32;
    const int tn   = blockIdx.y * 64  + (wave >> 2) * 32;
    const int half = lane >> 4, l16 = lane & 15;

    const int am0 = tm + l16, am1 = tm + 16 + l16;
    const int bn0 = tn + l16, bn1 = tn + 16 + l16;

    v8f c00 = {}, c01 = {}, c10 = {}, c11 = {};
    for (int k0 = 0; k0 < K; k0 += 32) {
        if (k0 + 32 < K) {
            __builtin_prefetch(A  + (size_t)am0 * K + k0 + 32, 0, 1);
            __builtin_prefetch(Bw + (size_t)bn0 * K + k0 + 32, 0, 1);
        }
        v16bf a0 = load_frag_bf(A,  K, am0, k0, half);
        v16bf a1 = load_frag_bf(A,  K, am1, k0, half);
        v16bf b0 = load_frag_bf(Bw, K, bn0, k0, half);
        v16bf b1 = load_frag_bf(Bw, K, bn1, k0, half);
        c00 = __builtin_amdgcn_wmma_f32_16x16x32_bf16(false, a0, false, b0, (short)0, c00, false, false);
        c01 = __builtin_amdgcn_wmma_f32_16x16x32_bf16(false, a0, false, b1, (short)0, c01, false, false);
        c10 = __builtin_amdgcn_wmma_f32_16x16x32_bf16(false, a1, false, b0, (short)0, c10, false, false);
        c11 = __builtin_amdgcn_wmma_f32_16x16x32_bf16(false, a1, false, b1, (short)0, c11, false, false);
    }

#pragma unroll
    for (int j = 0; j < 2; ++j) {
        const int gn = tn + j * 16 + l16;       // e = h*HD + hd
        const int h  = gn >> 6;
        const int hd = gn & 63;
        const float bv = bias[gn];
#pragma unroll
        for (int i = 0; i < 2; ++i) {
            const v8f& acc = (i == 0) ? (j == 0 ? c00 : c01)
                                      : (j == 0 ? c10 : c11);
#pragma unroll
            for (int v = 0; v < 8; ++v) {
                const int gm = tm + i * 16 + v + 8 * half;  // row = lv*BZ + b
                const int lv = gm >> 2;
                const int b  = gm & 3;
                const bool dead = mask[(size_t)b * LV_DIM + lv] != 0;
                const size_t n = (size_t)(b * H_DIM + h);
                Vs[(n * LV_DIM + lv) * HD_DIM + hd] = dead ? 0.0f : (acc[v] + bv);
            }
        }
    }
}

// ---------------------------------------------------------------------------
// Sampling: one wave per (n = b*H + h, q).  Softmax over P=4, linear interp
// gathers from Vs[n][lv][hd] (float2 per lane -> 256B coalesced per wave),
// writes attn_bf[(q*BZ+b)*E + h*HD + hd] directly in bf16 for the out-GEMM.
// ---------------------------------------------------------------------------
__global__ __launch_bounds__(256) void deform_sample_kernel(
    const float* __restrict__ Vs,
    const float* __restrict__ off_raw,   // [(q*BZ+b)*H*P + h*P + p]
    const float* __restrict__ aw_raw,    // same layout
    const float* __restrict__ ref_pt,    // [BZ, LQ]
    const float* __restrict__ snip,      // [BZ]
    __bf16* __restrict__ attn_bf)        // [LQ*BZ, E]
{
    const int w    = blockIdx.x * 8 + (threadIdx.x >> 5);
    const int lane = threadIdx.x & 31;
    const int q = w & (LQ_DIM - 1);
    const int n = w >> 11;               // LQ = 2048 = 2^11
    const int b = n >> 4;
    const int h = n & 15;
    const int hd0 = lane * 2;

    const float ref = ref_pt[(size_t)b * LQ_DIM + q];
    const float sn  = snip[b];
    const int   row = (q * BZ_DIM + b) * (H_DIM * P_DIM) + h * P_DIM;

    // softmax over P=4 (computed redundantly per lane)
    float a0 = aw_raw[row + 0], a1 = aw_raw[row + 1];
    float a2 = aw_raw[row + 2], a3 = aw_raw[row + 3];
    float mx = fmaxf(fmaxf(a0, a1), fmaxf(a2, a3));
    float e0 = __expf(a0 - mx), e1 = __expf(a1 - mx);
    float e2 = __expf(a2 - mx), e3 = __expf(a3 - mx);
    const float inv = 1.0f / (e0 + e1 + e2 + e3);
    float aw[P_DIM] = { e0 * inv, e1 * inv, e2 * inv, e3 * inv };

    const float* vbase = Vs + (size_t)n * LV_DIM * HD_DIM + hd0;
    float2 acc = make_float2(0.0f, 0.0f);

#pragma unroll
    for (int p = 0; p < P_DIM; ++p) {
        const float offv = off_raw[row + p];
        // grid = 2*(ref + off/snip) - 1; pos = (grid+1)*0.5*(L-1) = t*(L-1)
        const float pos = (ref + offv / sn) * (float)(LV_DIM - 1);
        const float i0f = floorf(pos);
        const float w1  = pos - i0f;
        const int i0 = (int)i0f;
        const int i1 = i0 + 1;

        float2 f0 = make_float2(0.0f, 0.0f), f1 = f0;
        if (i0 >= 0 && i0 < LV_DIM)
            f0 = *(const float2*)(vbase + (size_t)i0 * HD_DIM);
        if (i1 >= 0 && i1 < LV_DIM)
            f1 = *(const float2*)(vbase + (size_t)i1 * HD_DIM);

        const float w0 = 1.0f - w1;
        acc.x += aw[p] * (f0.x * w0 + f1.x * w1);
        acc.y += aw[p] * (f0.y * w0 + f1.y * w1);
    }

    __bf16* out = attn_bf + (size_t)(q * BZ_DIM + b) * E_DIM + h * HD_DIM + hd0;
    out[0] = (__bf16)acc.x;
    out[1] = (__bf16)acc.y;
}

// ---------------------------------------------------------------------------
extern "C" void kernel_launch(void* const* d_in, const int* in_sizes, int n_in,
                              void* d_out, int out_size, void* d_ws, size_t ws_size,
                              hipStream_t stream) {
    const float*         query  = (const float*)d_in[0];   // [LQ, BZ, E]
    const float*         value  = (const float*)d_in[1];   // [LV, BZ, E]
    const unsigned char* vmask  = (const unsigned char*)d_in[2]; // [BZ, LV] bool
    // d_in[3] value_valid_ratio: unused by reference
    const float*         refpt  = (const float*)d_in[4];   // [BZ, LQ, 1]
    const float*         snum   = (const float*)d_in[5];   // [BZ]
    const float*         Wv     = (const float*)d_in[6];   // [E, E]
    const float*         bv     = (const float*)d_in[7];   // [E]
    const float*         Woff   = (const float*)d_in[8];   // [H*P, E]
    const float*         boff   = (const float*)d_in[9];   // [H*P]
    const float*         Wattn  = (const float*)d_in[10];  // [H*P, E]
    const float*         battn  = (const float*)d_in[11];  // [H*P]
    const float*         Wout   = (const float*)d_in[12];  // [E, E]
    const float*         bout   = (const float*)d_in[13];  // [E]
    float*               out    = (float*)d_out;           // [LQ, BZ, E]

    const size_t MB = 1024 * 1024;
    char* ws = (char*)d_ws;
    // Workspace layout:
    float*  off_raw  = (float*) (ws);                  //  2 MiB
    float*  aw_raw   = (float*) (ws + 2   * MB);       //  2 MiB
    float*  Vs       = (float*) (ws + 4   * MB);       // 64 MiB
    __bf16* query_bf = (__bf16*)(ws + 68  * MB);       // 16 MiB
    __bf16* value_bf = (__bf16*)(ws + 84  * MB);       // 32 MiB
    __bf16* Wv_bf    = (__bf16*)(ws + 116 * MB);       //  2 MiB
    __bf16* Wout_bf  = (__bf16*)(ws + 118 * MB);       //  2 MiB
    __bf16* Woff_bf  = (__bf16*)(ws + 120 * MB);       // 128 KiB
    __bf16* Wattn_bf = (__bf16*)(ws + 120 * MB + 256 * 1024); // 128 KiB
    __bf16* attn_bf  = (__bf16*)(ws + 121 * MB);       // 16 MiB

    const dim3 blk(256);
    const int ELEMS_PER_BLOCK = 256 * 8;

    // 0) pack all GEMM operands to bf16 (one streaming pass each)
    pack_bf16_kernel<<<(LQ_DIM * BZ_DIM * E_DIM) / ELEMS_PER_BLOCK, blk, 0, stream>>>(query, query_bf);
    pack_bf16_kernel<<<(LV_DIM * BZ_DIM * E_DIM) / ELEMS_PER_BLOCK, blk, 0, stream>>>(value, value_bf);
    pack_bf16_kernel<<<(E_DIM * E_DIM) / ELEMS_PER_BLOCK, blk, 0, stream>>>(Wv,   Wv_bf);
    pack_bf16_kernel<<<(E_DIM * E_DIM) / ELEMS_PER_BLOCK, blk, 0, stream>>>(Wout, Wout_bf);
    pack_bf16_kernel<<<(H_DIM * P_DIM * E_DIM) / ELEMS_PER_BLOCK, blk, 0, stream>>>(Woff,  Woff_bf);
    pack_bf16_kernel<<<(H_DIM * P_DIM * E_DIM) / ELEMS_PER_BLOCK, blk, 0, stream>>>(Wattn, Wattn_bf);

    // 1) value projection -> masked, scattered into [bz*H, LV, HD]
    {
        dim3 grid((LV_DIM * BZ_DIM) / 128, E_DIM / 64);
        gemm_value_wmma<<<grid, blk, 0, stream>>>(value_bf, Wv_bf, bv, vmask, Vs);
    }
    // 2) sampling-offset projection: [LQ*BZ, H*P]
    {
        dim3 grid((LQ_DIM * BZ_DIM) / 128, (H_DIM * P_DIM) / 64);
        gemm_bias_wmma<<<grid, blk, 0, stream>>>(query_bf, Woff_bf, boff, off_raw,
                                                 LQ_DIM * BZ_DIM, H_DIM * P_DIM, E_DIM);
    }
    // 3) attention-weight projection: [LQ*BZ, H*P]
    {
        dim3 grid((LQ_DIM * BZ_DIM) / 128, (H_DIM * P_DIM) / 64);
        gemm_bias_wmma<<<grid, blk, 0, stream>>>(query_bf, Wattn_bf, battn, aw_raw,
                                                 LQ_DIM * BZ_DIM, H_DIM * P_DIM, E_DIM);
    }
    // 4) softmax + grid-sample + weighted sum -> attn_bf [LQ*BZ, E] (bf16)
    {
        dim3 grid((BZ_DIM * H_DIM * LQ_DIM) / 8);   // one wave per (n,q)
        deform_sample_kernel<<<grid, blk, 0, stream>>>(Vs, off_raw, aw_raw,
                                                       refpt, snum, attn_bf);
    }
    // 5) output projection -> d_out [LQ, BZ, E]
    {
        dim3 grid((LQ_DIM * BZ_DIM) / 128, E_DIM / 64);
        gemm_bias_wmma<<<grid, blk, 0, stream>>>(attn_bf, Wout_bf, bout, out,
                                                 LQ_DIM * BZ_DIM, E_DIM, E_DIM);
    }
}
